// Block_30605936951767
// MI455X (gfx1250) — matrix operand order repaired and checked by reading, not compiled, not run on hardware
//
#include <hip/hip_runtime.h>
#include <hip/hip_bf16.h>
#include <math.h>

// ---------------------------------------------------------------------------
// CDNA5 (gfx1250) wave32 WMMA transformer block.
//   D = A(16x32 f16) x B(32x16 f16) + C(16x16 f32)  via v_wmma_f32_16x16x32_f16
// Fragment layouts used here (contiguous K, same order for A and B):
//   A: lane<16 holds row (lane&15), K {k0..k0+15};
//      lane>=16 holds same row set, K {k0+16..k0+31}  -> two b128 loads
//   B: lane<16 holds col (lane&15), K {k0..k0+15};
//      lane>=16 holds col, K {k0+16..k0+31}                  -> one 32B load
//   C/D: lane<16 holds col, rows v=0..7; lane>=16 rows 8..15
// GEMM uses a 64x64 tile per wave (16 WMMA per 16 b128 loads per K-step).
// ---------------------------------------------------------------------------

typedef __attribute__((ext_vector_type(16))) _Float16 v16h;
typedef __attribute__((ext_vector_type(8)))  _Float16 v8h;
typedef __attribute__((ext_vector_type(8)))  float    v8f;

#define DEV static __device__ __forceinline__

DEV v8f v8f_zero() {
  v8f r;
#pragma unroll
  for (int i = 0; i < 8; ++i) r[i] = 0.0f;
  return r;
}

DEV v8f wmma_f16(v16h a, v16h b, v8f c) {
  return __builtin_amdgcn_wmma_f32_16x16x32_f16(
      /*neg_a=*/false, a, /*neg_b=*/false, b,
      /*c_mod=*/(short)0, c, /*reuse_a=*/false, /*reuse_b=*/false);
}

// A fragment: rows [row0, row0+16), K [k0, k0+32) from row-major f16 [*, lda]
DEV v16h load_a_frag(const _Float16* A, int lda, int row0, int k0, int lane) {
  int half = lane >> 4;
  const _Float16* p = A + (size_t)(row0 + (lane & 15)) * lda + k0 + half * 16;
  v8h lo = *(const v8h*)p;         // K k0+half*16 .. +7
  v8h hi = *(const v8h*)(p + 8);   // K k0+half*16+8 .. +15
  v16h r;
#pragma unroll
  for (int i = 0; i < 8; ++i) { r[i] = lo[i]; r[8 + i] = hi[i]; }
  return r;
}

// B fragment from weight stored [N][K] row-major: cols [n0,n0+16), K [k0,k0+32)
DEV v16h load_b_frag(const _Float16* W, int ldw, int n0, int k0, int lane) {
  const _Float16* p =
      W + (size_t)(n0 + (lane & 15)) * ldw + k0 + ((lane >> 4) << 4);
  return *(const v16h*)p;  // 32B, 32B-aligned -> two global_load_b128
}

// ---------------------------------------------------------------------------
// fp32 -> f16 convert
// ---------------------------------------------------------------------------
__global__ void f32_to_f16_kernel(const float* __restrict__ in,
                                  _Float16* __restrict__ out, int n) {
  int i = blockIdx.x * blockDim.x + threadIdx.x;
  if (i < n) out[i] = (_Float16)in[i];
}

// ---------------------------------------------------------------------------
// LayerNorm (ddof=1 variance, eps AFTER sqrt) -> f16 output. One row / block.
// ---------------------------------------------------------------------------
__global__ __launch_bounds__(256) void layernorm_f16_kernel(
    const float* __restrict__ x, const float* __restrict__ g,
    const float* __restrict__ be, _Float16* __restrict__ out) {
  const int NE = 1024;
  int row = blockIdx.x;
  int tid = threadIdx.x;
  const float* xr = x + (size_t)row * NE;
  float vals[4];
  float s = 0.f, s2 = 0.f;
#pragma unroll
  for (int i = 0; i < 4; ++i) {
    float v = xr[tid + i * 256];
    vals[i] = v;
    s += v;
    s2 += v * v;
  }
#pragma unroll
  for (int d = 1; d < 32; d <<= 1) {
    s += __shfl_xor(s, d, 32);
    s2 += __shfl_xor(s2, d, 32);
  }
  __shared__ float red0[8], red1[8];
  int w = tid >> 5;
  if ((tid & 31) == 0) { red0[w] = s; red1[w] = s2; }
  __syncthreads();
  float ts = 0.f, ts2 = 0.f;
#pragma unroll
  for (int i = 0; i < 8; ++i) { ts += red0[i]; ts2 += red1[i]; }
  float mu = ts * (1.0f / 1024.0f);
  float var = (ts2 - 1024.0f * mu * mu) * (1.0f / 1023.0f);
  float rs = 1.0f / (sqrtf(fmaxf(var, 0.0f)) + 1e-8f);
#pragma unroll
  for (int i = 0; i < 4; ++i) {
    int c = tid + i * 256;
    out[(size_t)row * NE + c] = (_Float16)(g[c] * ((vals[i] - mu) * rs) + be[c]);
  }
}

// ---------------------------------------------------------------------------
// Generic WMMA GEMM: C[M,N] = A[M,K](f16,row-major) * W[N,K](f16)^T + bias
// One wave per 64x64 output tile: 4 M-subtiles x 4 N-subtiles, so each A and
// each B fragment is reused 4x (16 WMMA : 16 b128 loads per K-step).
// ---------------------------------------------------------------------------
enum { EPI_F16 = 0, EPI_TRANS_V = 1, EPI_RES_F32 = 2, EPI_GELU_F16 = 3 };

template <int EPI>
__global__ __launch_bounds__(32) void gemm_wmma_kernel(
    const _Float16* __restrict__ A, const _Float16* __restrict__ W,
    const float* __restrict__ bias, const float* __restrict__ res,
    float* __restrict__ out32, _Float16* __restrict__ out16, int M, int N,
    int Kd) {
  int lane = threadIdx.x & 31;
  int ntn = N >> 6;
  int tn = blockIdx.x % ntn;
  int tm = blockIdx.x / ntn;
  int row0 = tm << 6;
  int n0 = tn << 6;

  v8f acc[4][4];
#pragma unroll
  for (int i = 0; i < 4; ++i)
#pragma unroll
    for (int j = 0; j < 4; ++j) acc[i][j] = v8f_zero();

  for (int k0 = 0; k0 < Kd; k0 += 32) {
    v16h a[4];
#pragma unroll
    for (int i = 0; i < 4; ++i)
      a[i] = load_a_frag(A, Kd, row0 + i * 16, k0, lane);
#pragma unroll
    for (int j = 0; j < 4; ++j) {
      v16h b = load_b_frag(W, Kd, n0 + j * 16, k0, lane);
#pragma unroll
      for (int i = 0; i < 4; ++i) acc[i][j] = wmma_f16(a[i], b, acc[i][j]);
    }
  }

  int n = lane & 15;
  int mb = (lane >> 4) * 8;
#pragma unroll
  for (int j = 0; j < 4; ++j) {
    int col = n0 + j * 16 + n;
    float bv = bias[col];
#pragma unroll
    for (int i = 0; i < 4; ++i) {
#pragma unroll
      for (int v = 0; v < 8; ++v) {
        int row = row0 + i * 16 + mb + v;
        float val = acc[i][j][v] + bv;
        if (EPI == EPI_GELU_F16)
          val = 0.5f * val * (1.0f + erff(val * 0.70710678118654752f));
        if (EPI == EPI_RES_F32) {
          size_t idx = (size_t)row * N + col;
          out32[idx] = val + res[idx];
        } else if (EPI == EPI_TRANS_V) {
          // store V transposed: Vt[b][h][d][s], S=2048, H=16, d_head=64
          int bb = row >> 11, sidx = row & 2047;
          int hh = col >> 6, d = col & 63;
          out16[((size_t)(bb * 16 + hh) * 64 + d) * 2048 + sidx] =
              (_Float16)val;
        } else {
          out16[(size_t)row * N + col] = (_Float16)val;
        }
      }
    }
  }
}

// ---------------------------------------------------------------------------
// Flash attention: one wave per (b, h, 16-row q tile). Causal + "tril==0"
// quirk. Online softmax; P staged through LDS to convert D-layout -> A-layout.
// ---------------------------------------------------------------------------
__global__ __launch_bounds__(32) void attn_kernel(
    const _Float16* __restrict__ Q, const _Float16* __restrict__ Km,
    const _Float16* __restrict__ Vt, _Float16* __restrict__ O) {
  const int S = 2048, H = 16, D = 64, NE = 1024;
  const float NEGB = -3.0e38f;
  int lane = threadIdx.x & 31;
  int qt = blockIdx.x & (S / 16 - 1);
  int bh = blockIdx.x / (S / 16);
  int h = bh & (H - 1);
  int b = bh / H;
  int q0 = qt * 16;

  const _Float16* Qb = Q + (size_t)b * S * NE + h * D;
  const _Float16* Kb = Km + (size_t)b * S * NE + h * D;
  const _Float16* Vb = Vt + (size_t)(b * H + h) * D * S;

  v16h qa0 = load_a_frag(Qb, NE, q0, 0, lane);
  v16h qa1 = load_a_frag(Qb, NE, q0, 32, lane);

  v8f oacc[4];
#pragma unroll
  for (int j = 0; j < 4; ++j) oacc[j] = v8f_zero();
  float mrow[8], lrow[8], alpha[8];
#pragma unroll
  for (int v = 0; v < 8; ++v) { mrow[v] = NEGB; lrow[v] = 0.0f; }

  __shared__ _Float16 Plds[16 * 32];
  int n = lane & 15;
  int mb = (lane >> 4) * 8;

  int kend = q0 + 16;  // causal: keys [0, q0+16)
  for (int kb = 0; kb < kend; kb += 32) {
    // ---- scores: two 16x16 tiles over K(head)=64 ----
    v8f sac[2];
#pragma unroll
    for (int t = 0; t < 2; ++t) {
      v8f a = v8f_zero();
      v16h b0 = load_b_frag(Kb, NE, kb + t * 16, 0, lane);
      v16h b1 = load_b_frag(Kb, NE, kb + t * 16, 32, lane);
      a = wmma_f16(qa0, b0, a);
      a = wmma_f16(qa1, b1, a);
      sac[t] = a;
    }
    // ---- scale, mask (causal + s==0 quirk), row max ----
    float mnew[8];
#pragma unroll
    for (int v = 0; v < 8; ++v) {
      int r = q0 + mb + v;
      float mx = mrow[v];
#pragma unroll
      for (int t = 0; t < 2; ++t) {
        float sc = sac[t][v] * 0.125f;  // / sqrt(64)
        int c = kb + t * 16 + n;
        if (c > r || sc == 0.0f) sc = NEGB;
        sac[t][v] = sc;
        mx = fmaxf(mx, sc);
      }
#pragma unroll
      for (int d = 1; d <= 8; d <<= 1)
        mx = fmaxf(mx, __shfl_xor(mx, d, 32));  // row lives in 16-lane half
      mnew[v] = mx;
    }
    // ---- exp / online rescale; stash P (f16) in LDS ----
#pragma unroll
    for (int v = 0; v < 8; ++v) {
      float al = __expf(mrow[v] - mnew[v]);  // underflows to 0 on first block
      mrow[v] = mnew[v];
      float rs = 0.0f;
#pragma unroll
      for (int t = 0; t < 2; ++t) {
        float sc = sac[t][v];
        float pe = (sc > -1.0e37f) ? __expf(sc - mnew[v]) : 0.0f;
        rs += pe;
        Plds[(mb + v) * 32 + t * 16 + n] = (_Float16)pe;
      }
#pragma unroll
      for (int d = 1; d <= 8; d <<= 1) rs += __shfl_xor(rs, d, 32);
      lrow[v] = lrow[v] * al + rs;
      alpha[v] = al;
    }
#pragma unroll
    for (int j = 0; j < 4; ++j)
#pragma unroll
      for (int v = 0; v < 8; ++v) oacc[j][v] *= alpha[v];
    __syncthreads();
    // ---- P(16x32) x V(32x64): A frag from LDS, B from Vt (contiguous) ----
    v16h pa = load_a_frag(Plds, 32, 0, 0, lane);
#pragma unroll
    for (int j = 0; j < 4; ++j) {
      v16h vb = load_b_frag(Vb, S, j * 16, kb, lane);
      oacc[j] = wmma_f16(pa, vb, oacc[j]);
    }
    __syncthreads();
  }
  // ---- normalize and store concat-head f16 output ----
#pragma unroll
  for (int j = 0; j < 4; ++j) {
#pragma unroll
    for (int v = 0; v < 8; ++v) {
      float val = oacc[j][v] / lrow[v];
      int r = q0 + mb + v;
      int c = h * 64 + j * 16 + n;
      O[(size_t)(b * S + r) * NE + c] = (_Float16)val;
    }
  }
}

// ---------------------------------------------------------------------------
// Host orchestration
// ---------------------------------------------------------------------------
extern "C" void kernel_launch(void* const* d_in, const int* in_sizes, int n_in,
                              void* d_out, int out_size, void* d_ws,
                              size_t ws_size, hipStream_t stream) {
  (void)in_sizes; (void)n_in; (void)out_size; (void)ws_size;
  const float* x   = (const float*)d_in[0];
  const float* Wq  = (const float*)d_in[1];
  const float* bq  = (const float*)d_in[2];
  const float* Wk  = (const float*)d_in[3];
  const float* bk  = (const float*)d_in[4];
  const float* Wv  = (const float*)d_in[5];
  const float* bv  = (const float*)d_in[6];
  const float* Wo  = (const float*)d_in[7];
  const float* bo  = (const float*)d_in[8];
  const float* g1  = (const float*)d_in[9];
  const float* be1 = (const float*)d_in[10];
  const float* g2  = (const float*)d_in[11];
  const float* be2 = (const float*)d_in[12];
  const float* W1  = (const float*)d_in[13];
  const float* bf1 = (const float*)d_in[14];
  const float* W2  = (const float*)d_in[15];
  const float* bf2 = (const float*)d_in[16];
  float* out = (float*)d_out;

  const int M = 4 * 2048, NE = 1024, NF = 4096, H = 16, S = 2048;

  char* ws = (char*)d_ws;
  size_t off = 0;
  auto a16 = [&](size_t e) { _Float16* p = (_Float16*)(ws + off); off += e * 2; return p; };
  auto a32 = [&](size_t e) { float* p = (float*)(ws + off); off += e * 4; return p; };
  _Float16* xn16 = a16((size_t)M * NE);
  _Float16* q16  = a16((size_t)M * NE);
  _Float16* k16  = a16((size_t)M * NE);
  _Float16* vt16 = a16((size_t)M * NE);  // transposed [b][h][d][s]
  _Float16* at16 = a16((size_t)M * NE);
  float*    x1   = a32((size_t)M * NE);
  _Float16* xn2  = a16((size_t)M * NE);
  _Float16* h16  = a16((size_t)M * NF);
  _Float16* wq16 = a16((size_t)NE * NE);
  _Float16* wk16 = a16((size_t)NE * NE);
  _Float16* wv16 = a16((size_t)NE * NE);
  _Float16* wo16 = a16((size_t)NE * NE);
  _Float16* w116 = a16((size_t)NF * NE);
  _Float16* w216 = a16((size_t)NE * NF);

  auto cvt = [&](const float* src, _Float16* dst, int n) {
    f32_to_f16_kernel<<<(n + 255) / 256, 256, 0, stream>>>(src, dst, n);
  };
  cvt(Wq, wq16, NE * NE);
  cvt(Wk, wk16, NE * NE);
  cvt(Wv, wv16, NE * NE);
  cvt(Wo, wo16, NE * NE);
  cvt(W1, w116, NF * NE);
  cvt(W2, w216, NE * NF);

  // LN1
  layernorm_f16_kernel<<<M, 256, 0, stream>>>(x, g1, be1, xn16);
  // QKV projections (weights are [N][K] row-major == WMMA B layout)
  int gProj = (M / 64) * (NE / 64);
  gemm_wmma_kernel<EPI_F16><<<gProj, 32, 0, stream>>>(
      xn16, wq16, bq, nullptr, nullptr, q16, M, NE, NE);
  gemm_wmma_kernel<EPI_F16><<<gProj, 32, 0, stream>>>(
      xn16, wk16, bk, nullptr, nullptr, k16, M, NE, NE);
  gemm_wmma_kernel<EPI_TRANS_V><<<gProj, 32, 0, stream>>>(
      xn16, wv16, bv, nullptr, nullptr, vt16, M, NE, NE);
  // attention
  attn_kernel<<<4 * H * (S / 16), 32, 0, stream>>>(q16, k16, vt16, at16);
  // output projection + residual (fp32)
  gemm_wmma_kernel<EPI_RES_F32><<<gProj, 32, 0, stream>>>(
      at16, wo16, bo, x, x1, nullptr, M, NE, NE);
  // LN2
  layernorm_f16_kernel<<<M, 256, 0, stream>>>(x1, g2, be2, xn2);
  // FFN1 + exact GELU
  gemm_wmma_kernel<EPI_GELU_F16><<<(M / 64) * (NF / 64), 32, 0, stream>>>(
      xn2, w116, bf1, nullptr, nullptr, h16, M, NF, NE);
  // FFN2 + residual -> final output
  gemm_wmma_kernel<EPI_RES_F32><<<gProj, 32, 0, stream>>>(
      h16, w216, bf2, x1, out, nullptr, M, NE, NF);
}
